// STGAT_73753178407547
// MI455X (gfx1250) — matrix operand order repaired and checked by reading, not compiled, not run on hardware
//
#include <hip/hip_runtime.h>
#include <cstdint>

// ---------------------------------------------------------------------------
// STGAT forward for MI455X (gfx1250, wave32).
// B=16, N=1024, NFEAT=16, NHID=32, NHEAD=4, NPRED=12
// ---------------------------------------------------------------------------

#define B_ 16
#define N_ 1024
#define NFEAT_ 16
#define NHID_ 32
#define NHEAD_ 4
#define NPRED_ 12
#define LRELU_A 0.2f

typedef _Float16 v16h __attribute__((ext_vector_type(16)));
typedef float    v8f  __attribute__((ext_vector_type(8)));

__device__ __forceinline__ float lane_read(float v, int srcLane) {
    int i = __builtin_amdgcn_ds_bpermute(srcLane << 2, __float_as_int(v));
    return __int_as_float(i);
}
__device__ __forceinline__ float xor16(float v) {
    int lane = threadIdx.x & 31;
    return lane_read(v, lane ^ 16);
}
__device__ __forceinline__ float lrelu(float v) { return v > 0.0f ? v : LRELU_A * v; }
__device__ __forceinline__ float elu(float v)  { return v > 0.0f ? v : (__expf(v) - 1.0f); }

// ---------------------------------------------------------------------------
// Kernel 1: pack adj [N,N] int32 -> bitmask [N][N/32] (128 KB, L2-resident)
// ---------------------------------------------------------------------------
__global__ void k_pack_adj(const int* __restrict__ adj, unsigned* __restrict__ bits) {
    int w = blockIdx.x * 256 + threadIdx.x;            // word index < N*N/32
    const int4* p = (const int4*)(adj + (size_t)w * 32);
    unsigned m = 0;
#pragma unroll
    for (int c = 0; c < 8; ++c) {
        int4 v = p[c];
        m |= (v.x > 0 ? 1u : 0u) << (c * 4 + 0);
        m |= (v.y > 0 ? 1u : 0u) << (c * 4 + 1);
        m |= (v.z > 0 ? 1u : 0u) << (c * 4 + 2);
        m |= (v.w > 0 ? 1u : 0u) << (c * 4 + 3);
    }
    bits[w] = m;
}

// ---------------------------------------------------------------------------
// Kernel 2: Wh = x @ W per (b,h); store WhT f16 [G][32][1024]; f_src/f_dst.
// grid (N/256, B*H), block 256
// ---------------------------------------------------------------------------
__global__ void k_proj1(const float* __restrict__ x, const float* __restrict__ W,
                        const float* __restrict__ a_src, const float* __restrict__ a_dst,
                        _Float16* __restrict__ VT, float* __restrict__ fsrc,
                        float* __restrict__ fdst) {
    int g = blockIdx.y;                  // g = b*NHEAD + h
    int b = g >> 2, h = g & 3;
    __shared__ float Ws[NFEAT_ * NHID_];
    __shared__ float as[NHID_], ad[NHID_];
    int tid = threadIdx.x;
    Ws[tid]       = W[(size_t)h * 512 + tid];
    Ws[tid + 256] = W[(size_t)h * 512 + tid + 256];
    if (tid < NHID_) { as[tid] = a_src[h * NHID_ + tid]; ad[tid] = a_dst[h * NHID_ + tid]; }
    __syncthreads();

    int n = blockIdx.x * 256 + tid;
    const float* xr = x + ((size_t)b * N_ + n) * NFEAT_;
    float xv[NFEAT_];
#pragma unroll
    for (int f = 0; f < NFEAT_; ++f) xv[f] = xr[f];

    float sfs = 0.f, sfd = 0.f;
#pragma unroll
    for (int k = 0; k < NHID_; ++k) {
        float acc = 0.f;
#pragma unroll
        for (int f = 0; f < NFEAT_; ++f) acc += xv[f] * Ws[f * NHID_ + k];
        VT[((size_t)g * NHID_ + k) * N_ + n] = (_Float16)acc;
        sfs += acc * as[k];
        sfd += acc * ad[k];
    }
    fsrc[(size_t)g * N_ + n] = sfs;
    fdst[(size_t)g * N_ + n] = sfd;
}

// ---------------------------------------------------------------------------
// Fused masked-softmax attention + WMMA P·V + ELU.
// NACC: # of 16-col f32 accumulators (2 for layer1 -> 32 cols, 1 for layer2).
// L1: store into head-concat buffer [B,N,128]; else store d_out [B,N,12].
// grid (N/128, G), block 256 (8 waves, 16 rows each).
// VT_all: [G][16*NACC][1024] f16 (row-major, K along fastest dim).
// ---------------------------------------------------------------------------
#define VSTRIDE 1032   // padded LDS row stride (halves) -> conflict mitigation

template <int NACC, bool L1>
__global__ void k_attn(const unsigned* __restrict__ adjbits,
                       const float* __restrict__ fsrc_all,
                       const float* __restrict__ fdst_all,
                       const _Float16* __restrict__ VT_all,
                       float* __restrict__ out) {
    __shared__ _Float16 vt[NACC * 16 * VSTRIDE];

    const int g   = blockIdx.y;
    const int tid = threadIdx.x;
    const int lane = tid & 31, laneLo = lane & 15, hi = lane >> 4;
    const int wave = tid >> 5;

    // --- stage V^T (f16) into LDS: NACC*16 rows x 1024 halves ---
    {
        const _Float16* src = VT_all + (size_t)g * (NACC * 16) * N_;
        const int chunks = NACC * 16 * 128;     // 8-half (16B) chunks
        for (int idx = tid; idx < chunks; idx += 256) {
            int row = idx >> 7, c8 = idx & 127;
            *(float4*)&vt[row * VSTRIDE + c8 * 8] =
                *(const float4*)&src[(size_t)row * N_ + c8 * 8];
        }
    }
    __syncthreads();

    const int row = blockIdx.x * 128 + wave * 16 + laneLo;   // global row this lane owns
    const float* fd = fdst_all + (size_t)g * N_;
    const float fs = fsrc_all[(size_t)g * N_ + row];

    // --- pass 1: analytic softmax max:  m = lrelu(fs + max_{masked j} fdst_j) ---
    float Md = -3.0e38f;
#pragma unroll 4
    for (int w = 0; w < 16; ++w) {
        unsigned word = adjbits[row * 32 + hi * 16 + w];
        while (word) {
            int bit = __builtin_ctz(word);
            word &= word - 1;
            Md = fmaxf(Md, fd[(hi * 16 + w) * 32 + bit]);
        }
    }
    Md = fmaxf(Md, xor16(Md));
    const float mrow = lrelu(fs + Md);

    // --- pass 2: flash loop, P tile in regs, WMMA f32 <- f16 ---
    v8f acc[NACC];
#pragma unroll
    for (int t = 0; t < NACC; ++t) acc[t] = (v8f){};
    float zsum = 0.f;
    const int khalf = hi * 8;       // A-fragment K offset (ISA 7.12.2)
    const int kbB   = hi * 8;       // B-fragment K offset (same K order as A)

    for (int jb = 0; jb < N_; jb += 32) {
        unsigned word = adjbits[row * 32 + (jb >> 5)];
        const float4 d0 = *(const float4*)(fd + jb + khalf);
        const float4 d1 = *(const float4*)(fd + jb + khalf + 4);
        const float4 d2 = *(const float4*)(fd + jb + khalf + 16);
        const float4 d3 = *(const float4*)(fd + jb + khalf + 20);
        const float dv[16] = {d0.x, d0.y, d0.z, d0.w, d1.x, d1.y, d1.z, d1.w,
                              d2.x, d2.y, d2.z, d2.w, d3.x, d3.y, d3.z, d3.w};
        v16h a;
#pragma unroll
        for (int i = 0; i < 16; ++i) {
            int k = khalf + i + (i >= 8 ? 8 : 0);     // lane<16: K{0..7,16..23}
            float e = lrelu(fs + dv[i]);
            float p = ((word >> k) & 1u) ? __expf(e - mrow) : 0.0f;
            _Float16 ph = (_Float16)p;
            a[i] = ph;
            zsum += (float)ph;
        }
#pragma unroll
        for (int t = 0; t < NACC; ++t) {
            const _Float16* bp = &vt[(t * 16 + laneLo) * VSTRIDE + jb + kbB];
            union { float4 f[2]; v16h v; } bu;
            bu.f[0] = *(const float4*)(bp);
            bu.f[1] = *(const float4*)(bp + 16);
            acc[t] = __builtin_amdgcn_wmma_f32_16x16x32_f16(
                false, a, false, bu.v, (short)0, acc[t], false, false);
        }
    }

    float ztot = zsum + xor16(zsum);
    ztot = fmaxf(ztot, 1e-30f);

    // --- normalize (single v_rcp_f32 per row), ELU, store.
    //     C/D layout: vgpr r = row r + 8*hi, col laneLo (ISA 7.12.2) ---
#pragma unroll
    for (int r = 0; r < 8; ++r) {
        const int n  = blockIdx.x * 128 + wave * 16 + r + 8 * hi;
        const float z  = lane_read(ztot, r + 8 * hi);
        const float rz = __builtin_amdgcn_rcpf(z);
#pragma unroll
        for (int t = 0; t < NACC; ++t) {
            float v = elu(acc[t][r] * rz);
            int c = t * 16 + laneLo;
            if (L1) {
                int b = g >> 2, h = g & 3;
                out[((size_t)b * N_ + n) * (NHEAD_ * NHID_) + h * NHID_ + c] = v;
            } else if (c < NPRED_) {
                out[((size_t)g * N_ + n) * NPRED_ + c] = v;
            }
        }
    }
}

// ---------------------------------------------------------------------------
// Kernel 4: Wh2 = h @ W_out ; g_src/g_dst ; store Wh2T f16 [B][16][1024]
// (cols 12..15 zero-padded). grid B*N/256, block 256.
// ---------------------------------------------------------------------------
__global__ void k_proj2(const float* __restrict__ hbuf, const float* __restrict__ Wout,
                        const float* __restrict__ aos, const float* __restrict__ aod,
                        _Float16* __restrict__ V2T, float* __restrict__ gsrc,
                        float* __restrict__ gdst) {
    __shared__ float Ws[128 * NPRED_];
    int tid = threadIdx.x;
    for (int i = tid; i < 128 * NPRED_; i += 256) Ws[i] = Wout[i];
    __syncthreads();

    int idx = blockIdx.x * 256 + tid;           // idx = b*N + n
    const float* hr = hbuf + (size_t)idx * 128;
    float s[NPRED_];
#pragma unroll
    for (int p = 0; p < NPRED_; ++p) s[p] = 0.f;
    for (int k = 0; k < 128; ++k) {
        float hv = hr[k];
#pragma unroll
        for (int p = 0; p < NPRED_; ++p) s[p] += hv * Ws[k * NPRED_ + p];
    }
    float gs = 0.f, gd = 0.f;
#pragma unroll
    for (int p = 0; p < NPRED_; ++p) { gs += s[p] * aos[p]; gd += s[p] * aod[p]; }
    int b = idx >> 10, n = idx & 1023;
#pragma unroll
    for (int p = 0; p < NPRED_; ++p)
        V2T[((size_t)b * 16 + p) * N_ + n] = (_Float16)s[p];
#pragma unroll
    for (int p = NPRED_; p < 16; ++p)
        V2T[((size_t)b * 16 + p) * N_ + n] = (_Float16)0.f;
    gsrc[idx] = gs;
    gdst[idx] = gd;
}

// ---------------------------------------------------------------------------
// launch
// ---------------------------------------------------------------------------
extern "C" void kernel_launch(void* const* d_in, const int* in_sizes, int n_in,
                              void* d_out, int out_size, void* d_ws, size_t ws_size,
                              hipStream_t stream) {
    (void)in_sizes; (void)n_in; (void)out_size; (void)ws_size;
    const float* x     = (const float*)d_in[0];
    const int*   adj   = (const int*)d_in[1];
    const float* W     = (const float*)d_in[2];
    const float* a_src = (const float*)d_in[3];
    const float* a_dst = (const float*)d_in[4];
    const float* W_out = (const float*)d_in[5];
    const float* a_os  = (const float*)d_in[6];
    const float* a_od  = (const float*)d_in[7];
    float* out = (float*)d_out;

    // workspace carve-up (all 256B aligned, ~13.9 MB total)
    char* ws = (char*)d_ws;
    unsigned*  adjbits = (unsigned*)(ws);                         // 131072 B
    float*     fsrc1   = (float*)(ws + 131072);                   // 262144 B
    float*     fdst1   = (float*)(ws + 393216);                   // 262144 B
    _Float16*  VT1     = (_Float16*)(ws + 655360);                // 4 MB
    float*     hbuf    = (float*)(ws + 4849664);                  // 8 MB
    float*     gsrc    = (float*)(ws + 13238272);                 // 64 KB
    float*     gdst    = (float*)(ws + 13303808);                 // 64 KB
    _Float16*  V2T     = (_Float16*)(ws + 13369344);              // 512 KB

    k_pack_adj<<<dim3(N_ * N_ / 32 / 256), dim3(256), 0, stream>>>(adj, adjbits);
    k_proj1<<<dim3(N_ / 256, B_ * NHEAD_), dim3(256), 0, stream>>>(
        x, W, a_src, a_dst, VT1, fsrc1, fdst1);
    k_attn<2, true><<<dim3(N_ / 128, B_ * NHEAD_), dim3(256), 0, stream>>>(
        adjbits, fsrc1, fdst1, VT1, hbuf);
    k_proj2<<<dim3(B_ * N_ / 256), dim3(256), 0, stream>>>(
        hbuf, W_out, a_os, a_od, V2T, gsrc, gdst);
    k_attn<1, false><<<dim3(N_ / 128, B_), dim3(256), 0, stream>>>(
        adjbits, gsrc, gdst, V2T, out);
}
